// VQVAE_41618233098591
// MI455X (gfx1250) — compile-verified, matrix operand order checked by reading, not run on hardware
//
#include <hip/hip_runtime.h>

// ---------------------------------------------------------------------------
// VQ-VAE forward on gfx1250: bf16 WMMA implicit-GEMM pipeline, double-buffered
// LDS with async-to-LDS staging (GLOBAL_LOAD_ASYNC_TO_LDS_B128, ASYNCcnt).
// ---------------------------------------------------------------------------

typedef unsigned short u16;
typedef __attribute__((ext_vector_type(16))) __bf16 v16bf;
typedef __attribute__((ext_vector_type(8)))  float  v8f;

#ifndef __has_builtin
#define __has_builtin(x) 0
#endif
#if __has_builtin(__builtin_amdgcn_global_load_async_to_lds_b128) && \
    __has_builtin(__builtin_amdgcn_s_wait_asynccnt)
#define USE_ASYNC_LDS 1
#else
#define USE_ASYNC_LDS 0
#endif

__device__ __forceinline__ float bf2f(u16 u) {
  union { unsigned int i; float f; } c; c.i = ((unsigned int)u) << 16; return c.f;
}
__device__ __forceinline__ u16 f2bf(float f) {
  union { float f; unsigned int i; } c; c.f = f;
  unsigned int x = c.i + 0x7FFFu + ((c.i >> 16) & 1u);   // round-to-nearest-even
  return (u16)(x >> 16);
}
// relu on two packed bf16 in one u32 (zero halves whose sign bit is set)
__device__ __forceinline__ unsigned int relu2(unsigned int x) {
  return x & ~(((x >> 15) & 0x00010001u) * 0xFFFFu);
}
__device__ __forceinline__ v8f wmma_bf16(v16bf a, v16bf b, v8f c) {
  return __builtin_amdgcn_wmma_f32_16x16x32_bf16(false, a, false, b, (short)0, c,
                                                 false, false);
}

// 16-byte global->LDS copy: async on CDNA5 (ASYNCcnt), else register round-trip
typedef int vi4_t __attribute__((vector_size(16)));
typedef __attribute__((address_space(1))) vi4_t* gp128_t;
typedef __attribute__((address_space(3))) vi4_t* lp128_t;

__device__ __forceinline__ void async_cp16(const u16* g, u16* l) {
#if USE_ASYNC_LDS
  __builtin_amdgcn_global_load_async_to_lds_b128((gp128_t)g, (lp128_t)l, 0, 0);
#else
  *(uint4*)l = *(const uint4*)g;
#endif
}
__device__ __forceinline__ void async_wait0() {
#if USE_ASYNC_LDS
  __builtin_amdgcn_s_wait_asynccnt(0);
#endif
}

// fragment = two 16B LDS reads from row-major [row][32k] tiles:
// lane slot i<8  -> k = klo + i ; slot i>=8 -> k = klo + 16 + (i-8); klo = 8*(lane>=16)
union FragU { uint4 q[2]; v16bf v; };

// ---------------------------------------------------------------------------
// Implicit-GEMM conv / convT.  MODE 0: conv (rows = output pixels).
// MODE 1: convT k4 s2 p1; blockIdx.z = output parity subgrid (4 taps).
// Workgroup 256 thr = 8 waves; tile 256(M) x 16*NTILES(N); wave owns 2 M-strips.
// Weights pre-transposed to bf16 [N][Ktot] (k-major) so B staging is a pure copy.
// ---------------------------------------------------------------------------
template<int MODE, int NTILES, bool RELU_IN, bool HAS_SKIP, bool OUT_F32>
__global__ __launch_bounds__(256, 2)
void gemm_conv_wmma(const u16* __restrict__ act,      // bf16 NHWC [BN][IH][IW][IC]
                    const u16* __restrict__ Bt,       // bf16 [N][Ktot]
                    const float* __restrict__ bias,
                    const u16* __restrict__ skip,     // bf16 NHWC or null
                    u16* __restrict__ out_bf,         // bf16 NHWC [BN][OH][OW][OC]
                    float* __restrict__ out_f32,      // f32 NCHW (final layer)
                    int BN, int IH, int IW, int IC,
                    int OH, int OW, int OC, int Ktot,
                    int KH, int KW, int stride, int pad,
                    int KB)
{
  const int tid = threadIdx.x, lane = tid & 31, wave = tid >> 5;
  const int m0 = blockIdx.x * 256;
  const int n0 = blockIdx.y * (NTILES * 16);
  int py = 0, px = 0;
  if (MODE == 1) { py = blockIdx.z >> 1; px = blockIdx.z & 1; }
  const int GH = (MODE == 0) ? OH : IH;
  const int GW = (MODE == 0) ? OW : IW;

  __shared__ __align__(32) u16 sA[2][256 * 32];      // 16KB x2
  __shared__ __align__(32) u16 sB[2][64 * 32];       //  4KB x2

  // A staging: this thread owns one M-row (32 channels = 64B per k-block)
  const int m   = m0 + tid;
  const int bb0 = m / (GH * GW);
  const int rm0 = m - bb0 * (GH * GW);
  const int yy  = rm0 / GW, xx = rm0 - yy * GW;
  const int CPB = IC >> 5;
  // B staging: thread copies one 16B chunk of an N-row
  const int bn = tid >> 2, bchunk = tid & 3;

  v8f acc[2][NTILES];
  #pragma unroll
  for (int st = 0; st < 2; ++st)
    #pragma unroll
    for (int j = 0; j < NTILES; ++j)
      #pragma unroll
      for (int e = 0; e < 8; ++e) acc[st][j][e] = 0.f;

  uint4 pq[4];            // in-flight A row (32 channels)
  bool  pvalid = false;

  auto prepare = [&](int kb, int buf) {
    const int tap = kb / CPB;
    const int c0  = (kb - tap * CPB) << 5;
    int dy, dx, kbase;
    if (MODE == 0) {
      const int r = tap / KW, s = tap - r * KW;
      dy = r - pad; dx = s - pad; kbase = tap * IC + c0;
    } else {
      const int ra = tap >> 1, sa = tap & 1;
      const int r = 2 * ra + (py ^ 1), s = 2 * sa + (px ^ 1);
      dy = py - ra; dx = px - sa; kbase = (r * 4 + s) * IC + c0;
    }
    // B: async copy (pure k-major copy)
    if (bn < 16 * NTILES) {
      const u16* src = Bt + (size_t)(n0 + bn) * Ktot + kbase + bchunk * 8;
      async_cp16(src, &sB[buf][bn * 32 + bchunk * 8]);
    }
    // A: issue global loads now, commit to LDS after compute
    const int iy = (MODE == 0) ? (yy * stride + dy) : (yy + dy);
    const int ix = (MODE == 0) ? (xx * stride + dx) : (xx + dx);
    pvalid = (iy >= 0 && iy < IH && ix >= 0 && ix < IW);
    if (pvalid) {
      const uint4* p4 = (const uint4*)(act + ((size_t)(bb0 * IH + iy) * IW + ix) * IC + c0);
      pq[0] = p4[0]; pq[1] = p4[1]; pq[2] = p4[2]; pq[3] = p4[3];
    }
  };
  auto commitA = [&](int buf) {
    if (pvalid) {
      if (RELU_IN) {
        unsigned int* u = (unsigned int*)pq;
      #pragma unroll
        for (int e = 0; e < 16; ++e) u[e] = relu2(u[e]);
      }
    } else {
      #pragma unroll
      for (int e = 0; e < 4; ++e) { pq[e].x = 0; pq[e].y = 0; pq[e].z = 0; pq[e].w = 0; }
    }
    uint4* d = (uint4*)&sA[buf][tid * 32];
    d[0] = pq[0]; d[1] = pq[1]; d[2] = pq[2]; d[3] = pq[3];
  };

  prepare(0, 0);
  commitA(0);
  const int klo = (lane >> 4) * 8;
  const int ln  = lane & 15;

  for (int kb = 0; kb < KB; ++kb) {
    const int buf = kb & 1;
    async_wait0();
    __syncthreads();
    const bool more = (kb + 1 < KB);
    if (more) prepare(kb + 1, buf ^ 1);       // overlap loads with compute
    // ---- compute from sA/sB[buf] ----
    FragU fa[2];
    #pragma unroll
    for (int st = 0; st < 2; ++st) {
      const int rowL = (wave * 2 + st) * 16 + ln;
      fa[st].q[0] = *(const uint4*)&sA[buf][rowL * 32 + klo];
      fa[st].q[1] = *(const uint4*)&sA[buf][rowL * 32 + klo + 16];
    }
    #pragma unroll
    for (int j = 0; j < NTILES; ++j) {
      FragU fb;
      const int rowN = j * 16 + ln;
      fb.q[0] = *(const uint4*)&sB[buf][rowN * 32 + klo];
      fb.q[1] = *(const uint4*)&sB[buf][rowN * 32 + klo + 16];
      acc[0][j] = wmma_bf16(fa[0].v, fb.v, acc[0][j]);
      acc[1][j] = wmma_bf16(fa[1].v, fb.v, acc[1][j]);
    }
    if (more) commitA(buf ^ 1);               // waits loads, stores to LDS
  }

  // ---- epilogue ----
  const int r8 = lane >> 4;
  #pragma unroll
  for (int st = 0; st < 2; ++st) {
    #pragma unroll
    for (int v = 0; v < 8; ++v) {
      const int mrow = m0 + (wave * 2 + st) * 16 + v + 8 * r8;
      const int bbo  = mrow / (GH * GW);
      const int rro  = mrow - bbo * (GH * GW);
      const int oy   = rro / GW, ox = rro - oy * GW;
      const int OY   = (MODE == 1) ? (2 * oy + py) : oy;
      const int OX   = (MODE == 1) ? (2 * ox + px) : ox;
      const size_t pix = (size_t)(bbo * OH + OY) * OW + OX;
      #pragma unroll
      for (int j = 0; j < NTILES; ++j) {
        const int col = n0 + j * 16 + ln;
        float val = acc[st][j][v];
        if (OUT_F32) {
          if (col < 3)
            out_f32[((size_t)(bbo * 3 + col) * OH + OY) * OW + OX] = val + bias[col];
        } else {
          val += bias[col];
          if (HAS_SKIP) val += bf2f(skip[pix * OC + col]);
          out_bf[pix * OC + col] = f2bf(val);
        }
      }
    }
  }
}

// ---------------------------------------------------------------------------
// VQ: argmin_c (|C_c|^2 - 2 z.C_c) via WMMA; fully async staging (pure copies).
// Tile 256 rows x full 512 codes (8 chunks of 64). Loss folds to
// (1+beta)*mean((z-zq)^2) because of the stop-gradients.
// ---------------------------------------------------------------------------
__global__ __launch_bounds__(256, 2)
void vq_argmin_kernel(const u16* __restrict__ ze,        // bf16 [65536][256]
                      const u16* __restrict__ cbBt,      // bf16 [512][256] k-major
                      const float* __restrict__ codebook,// f32 [512][256]
                      const float* __restrict__ cnorm,   // f32 [512]
                      u16* __restrict__ zq,
                      float* __restrict__ loss_accum)
{
  const int tid = threadIdx.x, lane = tid & 31, wave = tid >> 5;
  const int m0 = blockIdx.x * 256;
  __shared__ __align__(32) u16 sA[2][256 * 32];
  __shared__ __align__(32) u16 sB[2][64 * 32];
  __shared__ int   sh_ids[256];
  __shared__ float sh_red[256];

  const int bn = tid >> 2, bchunk = tid & 3;
  const int klo = (lane >> 4) * 8, ln = lane & 15;

  float minv[16]; int mini[16];
  #pragma unroll
  for (int i = 0; i < 16; ++i) { minv[i] = 3.4e38f; mini[i] = 0; }

  auto stage = [&](int step, int buf) {
    const int chunk = step >> 3, kb = step & 7;
    // A: thread copies its own row's 64B slice
    const u16* ap = ze + (size_t)(m0 + tid) * 256 + kb * 32;
    #pragma unroll
    for (int i = 0; i < 4; ++i)
      async_cp16(ap + i * 8, &sA[buf][tid * 32 + i * 8]);
    // B: 64 codebook rows x 64B
    const u16* bp = cbBt + (size_t)(chunk * 64 + bn) * 256 + kb * 32 + bchunk * 8;
    async_cp16(bp, &sB[buf][bn * 32 + bchunk * 8]);
  };

  v8f acc[2][4];
  #pragma unroll
  for (int st = 0; st < 2; ++st)
    #pragma unroll
    for (int j = 0; j < 4; ++j)
      #pragma unroll
      for (int e = 0; e < 8; ++e) acc[st][j][e] = 0.f;

  stage(0, 0);
  for (int step = 0; step < 64; ++step) {
    const int buf = step & 1;
    async_wait0();
    __syncthreads();
    if (step + 1 < 64) stage(step + 1, buf ^ 1);
    FragU fa[2];
    #pragma unroll
    for (int st = 0; st < 2; ++st) {
      const int rowL = (wave * 2 + st) * 16 + ln;
      fa[st].q[0] = *(const uint4*)&sA[buf][rowL * 32 + klo];
      fa[st].q[1] = *(const uint4*)&sA[buf][rowL * 32 + klo + 16];
    }
    #pragma unroll
    for (int j = 0; j < 4; ++j) {
      FragU fb;
      const int rowN = j * 16 + ln;
      fb.q[0] = *(const uint4*)&sB[buf][rowN * 32 + klo];
      fb.q[1] = *(const uint4*)&sB[buf][rowN * 32 + klo + 16];
      acc[0][j] = wmma_bf16(fa[0].v, fb.v, acc[0][j]);
      acc[1][j] = wmma_bf16(fa[1].v, fb.v, acc[1][j]);
    }
    if ((step & 7) == 7) {                    // end of K for this 64-code chunk
      const int chunk = step >> 3;
      #pragma unroll
      for (int st = 0; st < 2; ++st)
        #pragma unroll
        for (int j = 0; j < 4; ++j) {
          const int col = chunk * 64 + j * 16 + ln;
          const float cn = cnorm[col];
          #pragma unroll
          for (int v = 0; v < 8; ++v) {
            const float d = cn - 2.f * acc[st][j][v];
            if (d < minv[st * 8 + v]) { minv[st * 8 + v] = d; mini[st * 8 + v] = col; }
            acc[st][j][v] = 0.f;
          }
        }
    }
  }

  // reduce (min,argmin) across the 16 lanes sharing each row
  for (int off = 1; off < 16; off <<= 1) {
    #pragma unroll
    for (int i = 0; i < 16; ++i) {
      const float ov = __shfl_xor(minv[i], off, 32);
      const int   oi = __shfl_xor(mini[i], off, 32);
      if (ov < minv[i] || (ov == minv[i] && oi < mini[i])) { minv[i] = ov; mini[i] = oi; }
    }
  }
  if ((lane & 15) == 0) {
    #pragma unroll
    for (int st = 0; st < 2; ++st)
      #pragma unroll
      for (int v = 0; v < 8; ++v)
        sh_ids[(wave * 2 + st) * 16 + v + 8 * (lane >> 4)] = mini[st * 8 + v];
  }
  __syncthreads();

  // gather z_q (one thread per row) + squared-error accumulation
  float sum = 0.f;
  {
    const int id = sh_ids[tid];
    const size_t pix = (size_t)(m0 + tid) * 256;
    const float* cw = codebook + (size_t)id * 256;
    const u16* zp = ze + pix;
    u16* qp = zq + pix;
    for (int c = 0; c < 256; c += 4) {
      const float4 w4 = *(const float4*)(cw + c);
      const float d0 = bf2f(zp[c    ]) - w4.x;
      const float d1 = bf2f(zp[c + 1]) - w4.y;
      const float d2 = bf2f(zp[c + 2]) - w4.z;
      const float d3 = bf2f(zp[c + 3]) - w4.w;
      sum += d0 * d0 + d1 * d1 + d2 * d2 + d3 * d3;
      qp[c    ] = f2bf(w4.x); qp[c + 1] = f2bf(w4.y);
      qp[c + 2] = f2bf(w4.z); qp[c + 3] = f2bf(w4.w);
    }
  }
  sh_red[tid] = sum;
  __syncthreads();
  for (int s = 128; s > 0; s >>= 1) {
    if (tid < s) sh_red[tid] += sh_red[tid + s];
    __syncthreads();
  }
  if (tid == 0) atomicAdd(loss_accum, sh_red[0]);
}

// ---------------------------------------------------------------------------
// e1: 3->256 4x4 s2 p1 direct conv (K=48, tiny share of FLOPs).
// ---------------------------------------------------------------------------
__global__ __launch_bounds__(256)
void e1_conv_kernel(const float* __restrict__ x, const float* __restrict__ w,
                    const float* __restrict__ bias, u16* __restrict__ out)
{
  const int pixi = blockIdx.x;
  const int b  = pixi >> 14;
  const int rm = pixi & 16383;
  const int oy = rm >> 7, ox = rm & 127;
  __shared__ float sx[48];
  const int t = threadIdx.x;
  if (t < 48) {
    const int c = t >> 4, r = (t >> 2) & 3, s = t & 3;
    const int iy = 2 * oy + r - 1, ix = 2 * ox + s - 1;
    float v = 0.f;
    if (iy >= 0 && iy < 256 && ix >= 0 && ix < 256)
      v = x[((size_t)(b * 3 + c) * 256 + iy) * 256 + ix];
    sx[t] = v;
  }
  __syncthreads();
  const float* wr = w + (size_t)t * 48;
  float acc = bias[t];
  #pragma unroll
  for (int i = 0; i < 48; i += 4) {
    const float4 w4 = *(const float4*)(wr + i);
    acc += w4.x * sx[i] + w4.y * sx[i + 1] + w4.z * sx[i + 2] + w4.w * sx[i + 3];
  }
  out[(size_t)pixi * 256 + t] = f2bf(acc);
}

// ---------------------------------------------------------------------------
// Weight transform: OIHW f32 -> bf16 k-major Bt[n=o][k=(r*KW+s)*IC+c].
// ---------------------------------------------------------------------------
__global__ __launch_bounds__(256)
void wcvt_kernel(const float* __restrict__ w, u16* __restrict__ Bo,
                 int O, int IC, int KHW, int Ktot)
{
  const size_t t = (size_t)blockIdx.x * 256 + threadIdx.x;
  const size_t total = (size_t)O * IC * KHW;
  if (t >= total) return;
  const int o   = (int)(t / (IC * KHW));
  const int rem = (int)(t - (size_t)o * (IC * KHW));
  const int c   = rem / KHW;
  const int rs  = rem - c * KHW;
  Bo[(size_t)o * Ktot + rs * IC + c] = f2bf(w[t]);
}

__global__ void cnorm_kernel(const float* __restrict__ cb, float* __restrict__ cn)
{
  const int i = blockIdx.x * 64 + threadIdx.x;
  if (i >= 512) return;
  const float* r = cb + (size_t)i * 256;
  float s = 0.f;
  for (int c = 0; c < 256; c += 4) {
    const float4 v = *(const float4*)(r + c);
    s += v.x * v.x + v.y * v.y + v.z * v.z + v.w * v.w;
  }
  cn[i] = s;
}

__global__ void zero_u16_kernel(u16* p, int n) {
  const int i = blockIdx.x * 256 + threadIdx.x;
  if (i < n) p[i] = 0;
}
__global__ void zero_f32_kernel(float* p, int n) {
  const int i = blockIdx.x * 256 + threadIdx.x;
  if (i < n) p[i] = 0.f;
}
__global__ void loss_final_kernel(const float* __restrict__ s, float* __restrict__ out) {
  if (threadIdx.x == 0) out[0] = 1.25f * s[0] / 16777216.f;   // (1+beta)*mean
}

// ---------------------------------------------------------------------------
extern "C" void kernel_launch(void* const* d_in, const int* in_sizes, int n_in,
                              void* d_out, int out_size, void* d_ws, size_t ws_size,
                              hipStream_t stream)
{
  (void)in_sizes; (void)n_in; (void)out_size; (void)ws_size;
  const float* x    = (const float*)d_in[0];
  const float* e1w  = (const float*)d_in[1];  const float* e1b = (const float*)d_in[2];
  const float* e2w  = (const float*)d_in[3];  const float* e2b = (const float*)d_in[4];
  const float* cb   = (const float*)d_in[13];
  const float* u1w  = (const float*)d_in[22]; const float* u1b = (const float*)d_in[23];
  const float* u2w  = (const float*)d_in[24]; const float* u2b = (const float*)d_in[25];
  const float* aw[4]  = {(const float*)d_in[5], (const float*)d_in[9],
                         (const float*)d_in[14], (const float*)d_in[18]};
  const float* ab[4]  = {(const float*)d_in[6], (const float*)d_in[10],
                         (const float*)d_in[15], (const float*)d_in[19]};
  const float* bw[4]  = {(const float*)d_in[7], (const float*)d_in[11],
                         (const float*)d_in[16], (const float*)d_in[20]};
  const float* bbv[4] = {(const float*)d_in[8], (const float*)d_in[12],
                         (const float*)d_in[17], (const float*)d_in[21]};

  char* ws = (char*)d_ws;
  size_t off = 0;
  auto alloc = [&](size_t bytes) -> void* {
    void* p = ws + off;
    off = (off + bytes + 255) & ~(size_t)255;
    return p;
  };
  const size_t N128 = (size_t)16 * 128 * 128 * 256;
  const size_t N64  = (size_t)16 * 64 * 64 * 256;

  u16* actA = (u16*)alloc(N128 * 2);                  // e1 out / u1 out
  u16* X    = (u16*)alloc(N64 * 2);                   // encoder trunk
  u16* T    = (u16*)alloc(N64 * 2);                   // res inner
  u16* ZQ   = (u16*)alloc(N64 * 2);                   // quantized trunk
  u16* Be2  = (u16*)alloc((size_t)256 * 4096 * 2);
  u16* Ba[4], *Bb[4];
  for (int i = 0; i < 4; ++i) Ba[i] = (u16*)alloc((size_t)256 * 2304 * 2);
  for (int i = 0; i < 4; ++i) Bb[i] = (u16*)alloc((size_t)256 * 256 * 2);
  u16* Bcb  = (u16*)alloc((size_t)512 * 256 * 2);
  u16* Bu1  = (u16*)alloc((size_t)256 * 4096 * 2);
  u16* Bu2  = (u16*)alloc((size_t)16 * 4096 * 2);
  float* cn   = (float*)alloc(512 * 4);
  float* lsum = (float*)alloc(4);
  float* recon = (float*)d_out;
  float* loss_out = recon + (size_t)16 * 3 * 256 * 256;

  auto wgrid = [](size_t total) { return (unsigned)((total + 255) / 256); };
  // ---- weight transforms (k-major bf16) ----
  wcvt_kernel<<<wgrid(256 * 256 * 16), 256, 0, stream>>>(e2w, Be2, 256, 256, 16, 4096);
  for (int i = 0; i < 4; ++i) {
    wcvt_kernel<<<wgrid(256 * 256 * 9), 256, 0, stream>>>(aw[i], Ba[i], 256, 256, 9, 2304);
    wcvt_kernel<<<wgrid(256 * 256), 256, 0, stream>>>(bw[i], Bb[i], 256, 256, 1, 256);
  }
  wcvt_kernel<<<wgrid(512 * 256), 256, 0, stream>>>(cb, Bcb, 512, 256, 1, 256);
  wcvt_kernel<<<wgrid(256 * 256 * 16), 256, 0, stream>>>(u1w, Bu1, 256, 256, 16, 4096);
  zero_u16_kernel<<<wgrid(16 * 4096), 256, 0, stream>>>(Bu2, 16 * 4096);
  wcvt_kernel<<<wgrid(3 * 256 * 16), 256, 0, stream>>>(u2w, Bu2, 3, 256, 16, 4096);
  cnorm_kernel<<<8, 64, 0, stream>>>(cb, cn);
  zero_f32_kernel<<<1, 256, 0, stream>>>(lsum, 1);

  // ---- encoder ----
  e1_conv_kernel<<<16 * 128 * 128, 256, 0, stream>>>(x, e1w, e1b, actA);
  gemm_conv_wmma<0, 4, false, false, false><<<dim3(256, 4, 1), 256, 0, stream>>>(
      actA, Be2, e2b, nullptr, X, nullptr,
      16, 128, 128, 256, 64, 64, 256, 4096, 4, 4, 2, 1, 128);

  for (int i = 0; i < 2; ++i) {   // res1, res2 on X
    gemm_conv_wmma<0, 4, true, false, false><<<dim3(256, 4, 1), 256, 0, stream>>>(
        X, Ba[i], ab[i], nullptr, T, nullptr,
        16, 64, 64, 256, 64, 64, 256, 2304, 3, 3, 1, 1, 72);
    gemm_conv_wmma<0, 4, true, true, false><<<dim3(256, 4, 1), 256, 0, stream>>>(
        T, Bb[i], bbv[i], X, X, nullptr,
        16, 64, 64, 256, 64, 64, 256, 256, 1, 1, 1, 0, 8);
  }

  // ---- vector quantization ----
  vq_argmin_kernel<<<256, 256, 0, stream>>>(X, Bcb, cb, cn, ZQ, lsum);

  for (int i = 2; i < 4; ++i) {   // res3, res4 on ZQ
    gemm_conv_wmma<0, 4, true, false, false><<<dim3(256, 4, 1), 256, 0, stream>>>(
        ZQ, Ba[i], ab[i], nullptr, T, nullptr,
        16, 64, 64, 256, 64, 64, 256, 2304, 3, 3, 1, 1, 72);
    gemm_conv_wmma<0, 4, true, true, false><<<dim3(256, 4, 1), 256, 0, stream>>>(
        T, Bb[i], bbv[i], ZQ, ZQ, nullptr,
        16, 64, 64, 256, 64, 64, 256, 256, 1, 1, 1, 0, 8);
  }

  // ---- decoder convT ----
  gemm_conv_wmma<1, 4, false, false, false><<<dim3(256, 4, 4), 256, 0, stream>>>(
      ZQ, Bu1, u1b, nullptr, actA, nullptr,
      16, 64, 64, 256, 128, 128, 256, 4096, 4, 4, 0, 0, 32);
  gemm_conv_wmma<1, 1, false, false, true><<<dim3(1024, 1, 4), 256, 0, stream>>>(
      actA, Bu2, u2b, nullptr, nullptr, recon,
      16, 128, 128, 256, 256, 256, 16, 4096, 4, 4, 0, 0, 32);

  loss_final_kernel<<<1, 64, 0, stream>>>(lsum, loss_out);
}
